// GCN_eva_81329500717149
// MI455X (gfx1250) — matrix operand order repaired and checked by reading, not compiled, not run on hardware
//
#include <hip/hip_runtime.h>
#include <math.h>

// ---------------- problem constants (from reference) ----------------
#define NNODES 50000
#define NFEAT  512
#define NHID   256
#define NCLASS 40

typedef __attribute__((ext_vector_type(16))) _Float16 v16h;
typedef __attribute__((ext_vector_type(8)))  _Float16 v8h;
typedef __attribute__((ext_vector_type(8)))  float    v8f;
typedef __attribute__((ext_vector_type(4)))  float    v4f;
typedef __attribute__((ext_vector_type(4)))  _Float16 v4h;

// =====================================================================
// f32 -> f16 elementwise convert (vectorized x4; n divisible by 4 here)
// =====================================================================
__global__ __launch_bounds__(256) void cvt_f32_to_f16(
    const float* __restrict__ src, _Float16* __restrict__ dst, int n4)
{
    int i = blockIdx.x * 256 + threadIdx.x;
    if (i < n4) {
        v4f s = *(const v4f*)(src + (size_t)i * 4);
        v4h d;
        d.x = (_Float16)s.x; d.y = (_Float16)s.y;
        d.z = (_Float16)s.z; d.w = (_Float16)s.w;
        *(v4h*)(dst + (size_t)i * 4) = d;
    }
}

// =====================================================================
// Pack W[K,N] (f32, row-major) into WMMA B-fragment order (f16):
// For ntile t, k-step s, lane l (half = l>>4, col = t*16 + (l&15)):
//   frag element i (0..15) = W[s*32 + half*8 + (i&8) + i][col]
// stored contiguously: Bp[((t*ksteps + s)*32 + l)*16 + i]
// One wave per (ntile, kstep).
// =====================================================================
__global__ __launch_bounds__(32) void pack_w_f16(
    const float* __restrict__ W, _Float16* __restrict__ Bp,
    int K, int N, int ksteps)
{
    const int ntile = blockIdx.x;
    const int ks    = blockIdx.y;
    const int lane  = threadIdx.x;
    const int half  = lane >> 4;
    const int col   = ntile * 16 + (lane & 15);

    _Float16* __restrict__ dst =
        Bp + ((size_t)(ntile * ksteps + ks) * 32 + lane) * 16;
    const int krow = ks * 32 + half * 8;
    #pragma unroll
    for (int i = 0; i < 16; ++i) {
        dst[i] = (_Float16)W[(size_t)(krow + (i & 8) + i) * N + col];
    }
}

// =====================================================================
// GEMM with bias: C[M,N] = A[M,K](f16) @ W[K,N] + b[N]  (f32 accumulate)
// V_WMMA_F32_16X16X32_F16.  One wave computes a 16x64 strip (4 N-tiles)
// so the A fragment is reused 4x per load.  B comes from the packed
// fragment-order buffer (one contiguous 32B load per lane per tile).
// M % 16 == 0, N % 64 == 0, K % 32 == 0 (all true here).
// =====================================================================
__global__ __launch_bounds__(32) void gemm_bias_wmma_f16(
    const _Float16* __restrict__ A,     // [M,K] row-major f16
    const _Float16* __restrict__ Bp,    // packed fragments
    const float*    __restrict__ bias,  // [N]
    float*          __restrict__ C,     // [M,N] row-major f32
    int M, int K, int N)
{
    const int tm   = blockIdx.x;          // M/16 tile
    const int tg   = blockIdx.y;          // group of 4 N-tiles
    const int lane = threadIdx.x;
    const int half = lane >> 4;
    const int l    = lane & 15;
    const int ksteps = K >> 5;

    const _Float16* __restrict__ Arow = A + (size_t)(tm * 16 + l) * K;

    v8f acc0 = {}, acc1 = {}, acc2 = {}, acc3 = {};

    for (int ks = 0; ks < ksteps; ++ks) {
        const int k = ks * 32;
        // A 16x32 f16 fragment: lane -> row, two contiguous 16B chunks
        //   elems 0..7  = A[row][k + half*8 .. +7]
        //   elems 8..15 = A[row][k + 16 + half*8 .. +7]
        v8h a0 = *(const v8h*)(Arow + k + half * 8);
        v8h a1 = *(const v8h*)(Arow + k + 16 + half * 8);
        v16h a = __builtin_shufflevector(a0, a1,
                 0,1,2,3,4,5,6,7,8,9,10,11,12,13,14,15);

        const size_t bbase = ((size_t)(tg * 4) * ksteps + ks) * 32 + lane;
        const v16h b0 = *(const v16h*)(Bp + (bbase                      ) * 16);
        const v16h b1 = *(const v16h*)(Bp + (bbase + (size_t)ksteps * 32) * 16);
        const v16h b2 = *(const v16h*)(Bp + (bbase + (size_t)ksteps * 64) * 16);
        const v16h b3 = *(const v16h*)(Bp + (bbase + (size_t)ksteps * 96) * 16);

        acc0 = __builtin_amdgcn_wmma_f32_16x16x32_f16(false, a, false, b0,
                   (short)0, acc0, false, false);
        acc1 = __builtin_amdgcn_wmma_f32_16x16x32_f16(false, a, false, b1,
                   (short)0, acc1, false, false);
        acc2 = __builtin_amdgcn_wmma_f32_16x16x32_f16(false, a, false, b2,
                   (short)0, acc2, false, false);
        acc3 = __builtin_amdgcn_wmma_f32_16x16x32_f16(false, a, false, b3,
                   (short)0, acc3, false, false);
    }

    // C/D layout: VGPR j -> M = j + half*8 ; N = lane&15
    const int baseM = tm * 16 + half * 8;
    v8f accs[4] = {acc0, acc1, acc2, acc3};
    #pragma unroll
    for (int t = 0; t < 4; ++t) {
        const int col = (tg * 4 + t) * 16 + l;
        const float bv = bias[col];
        #pragma unroll
        for (int j = 0; j < 8; ++j) {
            C[(size_t)(baseM + j) * N + col] = accs[t][j] + bv;
        }
    }
}

// =====================================================================
// SpMM + ReLU:  O[r,:] = relu( sum_{e: rows[e]==r} vals[e] * H[cols[e],:] )
// edge_rows sorted ascending -> binary-search the per-row edge range.
// One 256-thread block per row; thread t owns feature column t.
// No atomics, fully deterministic.  Prefetch upcoming gathered row.
// =====================================================================
__device__ __forceinline__ int lower_bound_i32(const int* __restrict__ a,
                                               int n, int key)
{
    int lo = 0, hi = n;
    while (lo < hi) {
        int mid = (lo + hi) >> 1;
        if (a[mid] < key) lo = mid + 1; else hi = mid;
    }
    return lo;
}

__global__ __launch_bounds__(NHID) void spmm_relu(
    const int*   __restrict__ erows,   // [E] sorted
    const int*   __restrict__ ecols,   // [E]
    const float* __restrict__ evals,   // [E]
    const float* __restrict__ H,       // [NNODES, NHID]
    float*       __restrict__ O,       // [NNODES, NHID]
    int E)
{
    const int r = blockIdx.x;
    const int f = threadIdx.x;

    // block-uniform bounds (compiler scalarizes)
    const int lo = lower_bound_i32(erows, E, r);
    const int hi = lower_bound_i32(erows, E, r + 1);

    float acc = 0.0f;
    for (int e = lo; e < hi; ++e) {
        const int   c = ecols[e];      // uniform across block -> broadcast
        const float v = evals[e];
        if (e + 4 < hi) {              // prefetch a future gathered row
            const int cp = ecols[e + 4];
            __builtin_prefetch(&H[(size_t)cp * NHID + f], 0, 3);
        }
        acc = fmaf(v, H[(size_t)c * NHID + f], acc);
    }
    O[(size_t)r * NHID + f] = fmaxf(acc, 0.0f);
}

// =====================================================================
// Classifier: out[r,:] = log_softmax( Z[r,:] @ Wfc + bfc )
// One 64-thread block per row; lanes 0..39 each own one class.
// =====================================================================
__global__ __launch_bounds__(64) void fc_logsoftmax(
    const float* __restrict__ Z,       // [NNODES, NHID]
    const float* __restrict__ Wfc,     // [NHID, NCLASS]
    const float* __restrict__ bfc,     // [NCLASS]
    float*       __restrict__ out)     // [NNODES, NCLASS]
{
    const int row = blockIdx.x;
    const int t   = threadIdx.x;

    __shared__ float sv[64];
    __shared__ float s_lse;

    float acc = 0.0f;
    if (t < NCLASS) {
        acc = bfc[t];
        const float* __restrict__ zr = Z + (size_t)row * NHID;
        #pragma unroll 4
        for (int f = 0; f < NHID; ++f) {
            acc = fmaf(zr[f], Wfc[(size_t)f * NCLASS + t], acc);
        }
    }
    sv[t] = (t < NCLASS) ? acc : -INFINITY;
    __syncthreads();

    if (t == 0) {
        float m = sv[0];
        for (int i = 1; i < NCLASS; ++i) m = fmaxf(m, sv[i]);
        float s = 0.0f;
        for (int i = 0; i < NCLASS; ++i) s += expf(sv[i] - m);
        s_lse = m + logf(s);
    }
    __syncthreads();

    if (t < NCLASS) {
        out[(size_t)row * NCLASS + t] = acc - s_lse;
    }
}

// =====================================================================
// kernel_launch
// inputs (setup_inputs order):
//  0 x[N,512] f32 | 1 edge_rows[E] i32 | 2 edge_cols[E] i32 | 3 edge_vals[E] f32
//  4 W1[512,256]  | 5 b1[256] | 6 W2[256,256] | 7 b2[256]
//  8 Wfc[256,40]  | 9 bfc[40]
// output: [N,40] f32 log-probs
// workspace layout (all fully rewritten each call):
//   buf_a f32 [50000,256]  | buf_b f32 [50000,256]
//   a16   f16 [50000,512]  (x_f16, then reused for h1_f16)
//   pW1   f16 packed 512x256 | pW2 f16 packed 256x256
// =====================================================================
extern "C" void kernel_launch(void* const* d_in, const int* in_sizes, int n_in,
                              void* d_out, int out_size, void* d_ws, size_t ws_size,
                              hipStream_t stream)
{
    const float* x     = (const float*)d_in[0];
    const int*   erows = (const int*)  d_in[1];
    const int*   ecols = (const int*)  d_in[2];
    const float* evals = (const float*)d_in[3];
    const float* W1    = (const float*)d_in[4];
    const float* b1    = (const float*)d_in[5];
    const float* W2    = (const float*)d_in[6];
    const float* b2    = (const float*)d_in[7];
    const float* Wfc   = (const float*)d_in[8];
    const float* bfc   = (const float*)d_in[9];
    float*       out   = (float*)d_out;

    const int E = in_sizes[1];

    const size_t nh = (size_t)NNODES * NHID;      // 12.8M
    float*     buf_a = (float*)d_ws;
    float*     buf_b = buf_a + nh;
    _Float16*  a16   = (_Float16*)(buf_b + nh);   // 50000*512 f16
    _Float16*  pW1   = a16 + (size_t)NNODES * NFEAT;
    _Float16*  pW2   = pW1 + (size_t)NFEAT * NHID;

    // ---- pack weights into WMMA fragment order (f16) ----
    pack_w_f16<<<dim3(NHID / 16, NFEAT / 32), 32, 0, stream>>>(
        W1, pW1, NFEAT, NHID, NFEAT / 32);
    pack_w_f16<<<dim3(NHID / 16, NHID / 32), 32, 0, stream>>>(
        W2, pW2, NHID, NHID, NHID / 32);

    // ---- layer 1 ----
    {
        const int n4 = (NNODES * NFEAT) / 4;      // 6.4M
        cvt_f32_to_f16<<<(n4 + 255) / 256, 256, 0, stream>>>(x, a16, n4);
    }
    gemm_bias_wmma_f16<<<dim3(NNODES / 16, NHID / 64), 32, 0, stream>>>(
        a16, pW1, b1, buf_a, NNODES, NFEAT, NHID);
    spmm_relu<<<NNODES, NHID, 0, stream>>>(erows, ecols, evals, buf_a, buf_b, E);

    // ---- layer 2 ----
    {
        const int n4 = (NNODES * NHID) / 4;       // 3.2M
        cvt_f32_to_f16<<<(n4 + 255) / 256, 256, 0, stream>>>(buf_b, a16, n4);
    }
    gemm_bias_wmma_f16<<<dim3(NNODES / 16, NHID / 64), 32, 0, stream>>>(
        a16, pW2, b2, buf_a, NNODES, NHID, NHID);
    spmm_relu<<<NNODES, NHID, 0, stream>>>(erows, ecols, evals, buf_a, buf_b, E);

    // ---- classifier + log_softmax ----
    fc_logsoftmax<<<NNODES, 64, 0, stream>>>(buf_b, Wfc, bfc, out);
}
